// CorrGraphConvolution_72258529788240
// MI455X (gfx1250) — matrix operand order repaired and checked by reading, not compiled, not run on hardware
//
#include <hip/hip_runtime.h>
#include <hip/hip_bf16.h>

// ---------------------------------------------------------------------------
// CorrGraphConvolution for MI455X (gfx1250, wave32, WMMA).
// Big GEMMs use bf16x3-split v_wmma_f32_16x16x32_bf16 (near-fp32 accuracy),
// software-pipelined: register prefetch of next k-tile + double-buffered LDS
// (one barrier per k-step). Small per-batch 14x14 graph ops + BatchNorm are
// VALU two-pass kernels.
// ---------------------------------------------------------------------------

typedef __bf16 bf16_t;
typedef __attribute__((ext_vector_type(4)))  __bf16 v4bf;
typedef __attribute__((ext_vector_type(8)))  __bf16 v8bf;
typedef __attribute__((ext_vector_type(16))) __bf16 v16bf;
typedef __attribute__((ext_vector_type(4)))  float  v4f;
typedef __attribute__((ext_vector_type(8)))  float  v8f;

#define NB    14
#define BATCH 4096
#define FIN   1024
#define FOUT  1024
#define HMID  512
#define MROWS (BATCH * NB)   // 57344
#define LSTR  40             // LDS row stride in bf16 (32 data + 8 pad = 80B)

__device__ __forceinline__ float waveReduceSum(float v) {
  #pragma unroll
  for (int m = 16; m > 0; m >>= 1) v += __shfl_xor(v, m, 32);
  return v;
}

__device__ __forceinline__ v16bf cat8(v8bf a, v8bf b) {
  return __builtin_shufflevector(a, b, 0,1,2,3,4,5,6,7,8,9,10,11,12,13,14,15);
}

// ---------------------------------------------------------------------------
// GEMM: C[M,Ncol] = A[M,K] @ W[K,Ncol]  (+ bias, leaky if epi==1)
// fp32 in/out, bf16x3 split WMMA accumulation in fp32.
// Block tile 128x128, K-tile 32, 256 threads (8 waves, 32x64 per wave).
// Pipelined: prefetch k-tile (kt+1) into registers while computing kt;
// double-buffered LDS -> one barrier per iteration.
// M % 128 == 0, Ncol % 128 == 0, K % 32 == 0 guaranteed by problem sizes.
// ---------------------------------------------------------------------------
__global__ __launch_bounds__(256)
void gemm_bf16x3_kernel(const float* __restrict__ A, const float* __restrict__ W,
                        const float* __restrict__ bias, float* __restrict__ C,
                        int M, int K, int Ncol, int epi)
{
  __shared__ bf16_t sAhi[2][128 * LSTR];
  __shared__ bf16_t sAlo[2][128 * LSTR];
  __shared__ bf16_t sBhi[2][128 * LSTR];   // transposed: [col][k]
  __shared__ bf16_t sBlo[2][128 * LSTR];

  const int tid   = threadIdx.x;
  const int lane  = tid & 31;
  const int wave  = tid >> 5;
  const int wm    = wave >> 1;     // 0..3  -> 32-row band
  const int wn    = wave & 1;      // 0..1  -> 64-col band
  const int m0    = blockIdx.y * 128;
  const int n0    = blockIdx.x * 128;
  const int khalf = lane >> 4;     // 0/1 half-lane group
  const int l16   = lane & 15;

  // staging coordinates (iteration 'it' shifts rows/k-rows only)
  const int rA0 = tid >> 3;          // A row base (+32 per it), 8 float4/row
  const int cA  = (tid & 7) << 2;    // A col within k-tile
  const int kB0 = tid >> 5;          // W k-row base (+8 per it), 32 float4/row
  const int cB  = (tid & 31) << 2;   // W col within n-tile

  v4f pa[4], pb[4];                  // register prefetch buffers

  auto loadG = [&](int kt) {
    #pragma unroll
    for (int it = 0; it < 4; ++it)
      pa[it] = *(const v4f*)(A + (size_t)(m0 + rA0 + it * 32) * K + kt * 32 + cA);
    #pragma unroll
    for (int it = 0; it < 4; ++it)
      pb[it] = *(const v4f*)(W + (size_t)(kt * 32 + kB0 + it * 8) * Ncol + n0 + cB);
  };

  auto stage = [&](int buf) {
    #pragma unroll
    for (int it = 0; it < 4; ++it) {
      v4bf h, l;
      #pragma unroll
      for (int e = 0; e < 4; ++e) {
        float f = pa[it][e];
        __bf16 hb = (__bf16)f;
        h[e] = hb;
        l[e] = (__bf16)(f - (float)hb);
      }
      *(v4bf*)&sAhi[buf][(rA0 + it * 32) * LSTR + cA] = h;
      *(v4bf*)&sAlo[buf][(rA0 + it * 32) * LSTR + cA] = l;
    }
    #pragma unroll
    for (int it = 0; it < 4; ++it) {
      #pragma unroll
      for (int e = 0; e < 4; ++e) {
        float f = pb[it][e];
        __bf16 hb = (__bf16)f;
        sBhi[buf][(cB + e) * LSTR + kB0 + it * 8] = hb;
        sBlo[buf][(cB + e) * LSTR + kB0 + it * 8] = (__bf16)(f - (float)hb);
      }
    }
  };

  v8f acc[2][4] = {};

  loadG(0);
  stage(0);

  const int ktiles = K >> 5;
  for (int kt = 0; kt < ktiles; ++kt) {
    const int buf = kt & 1;
    if (kt + 1 < ktiles) loadG(kt + 1);   // prefetch next tile (hidden by WMMAs)
    __syncthreads();                      // LDS writes of 'buf' visible

    // ---- load fragments (layouts per CDNA5 ISA 7.12.2) ----
    v16bf ah[2], al[2];
    #pragma unroll
    for (int i = 0; i < 2; ++i) {
      int row = wm * 32 + i * 16 + l16;
      const bf16_t* ph = &sAhi[buf][row * LSTR + khalf * 8];
      const bf16_t* pl = &sAlo[buf][row * LSTR + khalf * 8];
      ah[i] = cat8(*(const v8bf*)ph, *(const v8bf*)(ph + 16));
      al[i] = cat8(*(const v8bf*)pl, *(const v8bf*)(pl + 16));
    }
    v16bf bh[4], bl[4];
    #pragma unroll
    for (int j = 0; j < 4; ++j) {
      int col = wn * 64 + j * 16 + l16;
      const bf16_t* ph = &sBhi[buf][col * LSTR + khalf * 8];
      const bf16_t* pl = &sBlo[buf][col * LSTR + khalf * 8];
      bh[j] = cat8(*(const v8bf*)ph, *(const v8bf*)(ph + 16));
      bl[j] = cat8(*(const v8bf*)pl, *(const v8bf*)(pl + 16));
    }

    // ---- bf16x3: hi*hi + hi*lo + lo*hi ----
    #pragma unroll
    for (int i = 0; i < 2; ++i) {
      #pragma unroll
      for (int j = 0; j < 4; ++j) {
        acc[i][j] = __builtin_amdgcn_wmma_f32_16x16x32_bf16(
            false, ah[i], false, bh[j], (short)0, acc[i][j], false, false);
        acc[i][j] = __builtin_amdgcn_wmma_f32_16x16x32_bf16(
            false, ah[i], false, bl[j], (short)0, acc[i][j], false, false);
        acc[i][j] = __builtin_amdgcn_wmma_f32_16x16x32_bf16(
            false, al[i], false, bh[j], (short)0, acc[i][j], false, false);
      }
    }

    // stage next tile into the other buffer; safe without extra barrier:
    // every wave writing buf^1 already passed this iteration's barrier,
    // which ordered it after all (kt-1) reads of buf^1.
    if (kt + 1 < ktiles) stage(buf ^ 1);
  }

  // ---- epilogue + store (C/D layout: row = v + 8*khalf, col = l16) ----
  #pragma unroll
  for (int i = 0; i < 2; ++i) {
    int rbase = m0 + wm * 32 + i * 16 + khalf * 8;
    #pragma unroll
    for (int j = 0; j < 4; ++j) {
      int gc = n0 + wn * 64 + j * 16 + l16;
      float bv = epi ? bias[gc] : 0.f;
      #pragma unroll
      for (int v = 0; v < 8; ++v) {
        float val = acc[i][j][v] + bv;
        if (epi) val = (val >= 0.f) ? val : 0.2f * val;
        C[(size_t)(rbase + v) * Ncol + gc] = val;
      }
    }
  }
}

// ---------------------------------------------------------------------------
// corr: abs cosine similarity of h rows per batch, diag = 1.
// ---------------------------------------------------------------------------
__global__ __launch_bounds__(256)
void corr_kernel(const float* __restrict__ h, float* __restrict__ corr)
{
  __shared__ float sh[NB * FOUT];
  __shared__ float rs[NB];
  const int b = blockIdx.x, tid = threadIdx.x, lane = tid & 31, wave = tid >> 5;
  const float* hb = h + (size_t)b * NB * FOUT;
  for (int i = tid; i < NB * FOUT / 4; i += 256)
    *(v4f*)&sh[i * 4] = *(const v4f*)&hb[i * 4];
  __syncthreads();
  for (int r = wave; r < NB; r += 8) {
    float s = 0.f;
    for (int c = lane; c < FOUT; c += 32) { float v = sh[r * FOUT + c]; s += v * v; }
    s = waveReduceSum(s);
    if (lane == 0) rs[r] = rsqrtf(fmaxf(s, 1e-8f * 1e-8f));
  }
  __syncthreads();
  float* cb = corr + (size_t)b * NB * NB;
  for (int p = wave; p < 91; p += 8) {          // 14*13/2 off-diag pairs
    int i = 0, pp = p;
    while (pp >= 13 - i) { pp -= 13 - i; ++i; }
    int j = i + 1 + pp;
    float d = 0.f;
    for (int c = lane; c < FOUT; c += 32) d += sh[i * FOUT + c] * sh[j * FOUT + c];
    d = waveReduceSum(d);
    if (lane == 0) {
      float v = fabsf(d) * rs[i] * rs[j];
      cb[i * NB + j] = v; cb[j * NB + i] = v;
    }
  }
  if (tid < NB) cb[tid * NB + tid] = 1.f;
}

// ---------------------------------------------------------------------------
// adjacency sym-normalization (diag forced to 1, d = clip(rowsum,1)^-0.5)
// ---------------------------------------------------------------------------
__global__ void adjnorm_kernel(const float* __restrict__ adj1,
                               const float* __restrict__ adj2,
                               float* __restrict__ adjn)
{
  __shared__ float sa[NB * NB];
  __shared__ float d[NB];
  const int t = threadIdx.x;
  for (int a = 0; a < 2; ++a) {
    const float* src = a ? adj2 : adj1;
    if (t < NB * NB) { int i = t / NB; sa[t] = (i == t % NB) ? 1.f : src[t]; }
    __syncthreads();
    if (t < NB) { float s = 0.f; for (int j = 0; j < NB; ++j) s += sa[t * NB + j];
                  d[t] = rsqrtf(fmaxf(s, 1.f)); }
    __syncthreads();
    if (t < NB * NB) adjn[a * NB * NB + t] = d[t / NB] * sa[t] * d[t % NB];
    __syncthreads();
  }
}

// ---------------------------------------------------------------------------
// gcn1 BN stats: y = adjn{1,2} @ xw1 + b1 ; accumulate sum/sumsq per node.
// stats layout: [s1(14), q1(14), s3(14), q3(14)]
// ---------------------------------------------------------------------------
__global__ __launch_bounds__(256)
void gcn1_stats_kernel(const float* __restrict__ xw, const float* __restrict__ adjn,
                       const float* __restrict__ b1, float* __restrict__ stats)
{
  __shared__ float sx[NB * HMID];
  __shared__ float sadj[2 * NB * NB];
  __shared__ float red[4 * NB];
  const int b = blockIdx.x, tid = threadIdx.x, lane = tid & 31;
  const float* xb = xw + (size_t)b * NB * HMID;
  for (int i = tid; i < NB * HMID / 4; i += 256)
    *(v4f*)&sx[i * 4] = *(const v4f*)&xb[i * 4];
  if (tid < 2 * NB * NB) sadj[tid] = adjn[tid];
  if (tid < 4 * NB) red[tid] = 0.f;
  __syncthreads();
  float s1[NB], q1[NB], s3[NB], q3[NB];
  #pragma unroll
  for (int n = 0; n < NB; ++n) { s1[n] = q1[n] = s3[n] = q3[n] = 0.f; }
  for (int c = tid; c < HMID; c += 256) {
    float bias = b1[c];
    #pragma unroll
    for (int n = 0; n < NB; ++n) {
      float y1 = bias, y3 = bias;
      #pragma unroll
      for (int m = 0; m < NB; ++m) {
        float xv = sx[m * HMID + c];
        y1 += sadj[n * NB + m] * xv;
        y3 += sadj[NB * NB + n * NB + m] * xv;
      }
      s1[n] += y1; q1[n] += y1 * y1; s3[n] += y3; q3[n] += y3 * y3;
    }
  }
  #pragma unroll
  for (int n = 0; n < NB; ++n) {
    float a = waveReduceSum(s1[n]), bq = waveReduceSum(q1[n]);
    float cc = waveReduceSum(s3[n]), dd = waveReduceSum(q3[n]);
    if (lane == 0) {
      atomicAdd(&red[n], a); atomicAdd(&red[NB + n], bq);
      atomicAdd(&red[2 * NB + n], cc); atomicAdd(&red[3 * NB + n], dd);
    }
  }
  __syncthreads();
  if (tid < 4 * NB) atomicAdd(&stats[tid], red[tid]);
}

// ---------------------------------------------------------------------------
// gcn1 apply: o = leaky(BN(adjn_p @ xw1 + b1; gamma,beta; global stats))
// ---------------------------------------------------------------------------
__global__ __launch_bounds__(256)
void gcn1_apply_kernel(const float* __restrict__ xw, const float* __restrict__ adjn_p,
                       const float* __restrict__ b1, const float* __restrict__ gamma,
                       const float* __restrict__ beta, const float* __restrict__ ssum,
                       const float* __restrict__ ssq, float* __restrict__ o)
{
  __shared__ float sx[NB * HMID];
  __shared__ float sadj[NB * NB];
  __shared__ float scale[NB], shift[NB];
  const int b = blockIdx.x, tid = threadIdx.x;
  const float* xb = xw + (size_t)b * NB * HMID;
  for (int i = tid; i < NB * HMID / 4; i += 256)
    *(v4f*)&sx[i * 4] = *(const v4f*)&xb[i * 4];
  if (tid < NB * NB) sadj[tid] = adjn_p[tid];
  if (tid < NB) {
    const float cnt = (float)BATCH * (float)HMID;
    float mu  = ssum[tid] / cnt;
    float var = ssq[tid] / cnt - mu * mu;
    float sc  = rsqrtf(var + 1e-5f) * gamma[tid];
    scale[tid] = sc; shift[tid] = beta[tid] - mu * sc;
  }
  __syncthreads();
  float* ob = o + (size_t)b * NB * HMID;
  for (int c = tid; c < HMID; c += 256) {
    float bias = b1[c];
    #pragma unroll
    for (int n = 0; n < NB; ++n) {
      float y = bias;
      #pragma unroll
      for (int m = 0; m < NB; ++m) y += sadj[n * NB + m] * sx[m * HMID + c];
      y = y * scale[n] + shift[n];
      ob[n * HMID + c] = (y >= 0.f) ? y : 0.2f * y;
    }
  }
}

// ---------------------------------------------------------------------------
// gcn2 BN stats: u = corrn @ z + b2 ; accumulate sum/sumsq per node.
// ---------------------------------------------------------------------------
__global__ __launch_bounds__(256)
void gcn2_stats_kernel(const float* __restrict__ z, const float* __restrict__ corr,
                       const float* __restrict__ b2, float* __restrict__ stats)
{
  __shared__ float sz[NB * FOUT];
  __shared__ float scn[NB * NB];
  __shared__ float dd[NB];
  __shared__ float red[2 * NB];
  const int b = blockIdx.x, tid = threadIdx.x, lane = tid & 31;
  const float* zb = z + (size_t)b * NB * FOUT;
  for (int i = tid; i < NB * FOUT / 4; i += 256)
    *(v4f*)&sz[i * 4] = *(const v4f*)&zb[i * 4];
  const float* cb = corr + (size_t)b * NB * NB;
  if (tid < NB * NB) scn[tid] = cb[tid];
  if (tid < 2 * NB) red[tid] = 0.f;
  __syncthreads();
  if (tid < NB) { float s = 0.f; for (int j = 0; j < NB; ++j) s += scn[tid * NB + j];
                  dd[tid] = rsqrtf(fmaxf(s, 1.f)); }
  __syncthreads();
  if (tid < NB * NB) scn[tid] = dd[tid / NB] * scn[tid] * dd[tid % NB];
  __syncthreads();
  float su[NB], qu[NB];
  #pragma unroll
  for (int n = 0; n < NB; ++n) { su[n] = qu[n] = 0.f; }
  for (int c = tid; c < FOUT; c += 256) {
    float bias = b2[c];
    #pragma unroll
    for (int n = 0; n < NB; ++n) {
      float u = bias;
      #pragma unroll
      for (int m = 0; m < NB; ++m) u += scn[n * NB + m] * sz[m * FOUT + c];
      su[n] += u; qu[n] += u * u;
    }
  }
  #pragma unroll
  for (int n = 0; n < NB; ++n) {
    float a = waveReduceSum(su[n]), q = waveReduceSum(qu[n]);
    if (lane == 0) { atomicAdd(&red[n], a); atomicAdd(&red[NB + n], q); }
  }
  __syncthreads();
  if (tid < 2 * NB) atomicAdd(&stats[tid], red[tid]);
}

// ---------------------------------------------------------------------------
// gcn2 final: out{2,4} = leaky(BN(corrn @ z + b2)); store or accumulate.
// ---------------------------------------------------------------------------
__global__ __launch_bounds__(256)
void gcn2_final_kernel(const float* __restrict__ z, const float* __restrict__ corr,
                       const float* __restrict__ b2, const float* __restrict__ gamma,
                       const float* __restrict__ beta, const float* __restrict__ stats,
                       float* __restrict__ outp, int add)
{
  __shared__ float sz[NB * FOUT];
  __shared__ float scn[NB * NB];
  __shared__ float dd[NB];
  __shared__ float scale[NB], shift[NB];
  const int b = blockIdx.x, tid = threadIdx.x;
  const float* zb = z + (size_t)b * NB * FOUT;
  for (int i = tid; i < NB * FOUT / 4; i += 256)
    *(v4f*)&sz[i * 4] = *(const v4f*)&zb[i * 4];
  const float* cb = corr + (size_t)b * NB * NB;
  if (tid < NB * NB) scn[tid] = cb[tid];
  if (tid < NB) {
    const float cnt = (float)BATCH * (float)FOUT;
    float mu  = stats[tid] / cnt;
    float var = stats[NB + tid] / cnt - mu * mu;
    float sc  = rsqrtf(var + 1e-5f) * gamma[tid];
    scale[tid] = sc; shift[tid] = beta[tid] - mu * sc;
  }
  __syncthreads();
  if (tid < NB) { float s = 0.f; for (int j = 0; j < NB; ++j) s += scn[tid * NB + j];
                  dd[tid] = rsqrtf(fmaxf(s, 1.f)); }
  __syncthreads();
  if (tid < NB * NB) scn[tid] = dd[tid / NB] * scn[tid] * dd[tid % NB];
  __syncthreads();
  float* ob = outp + (size_t)b * NB * FOUT;
  for (int c = tid; c < FOUT; c += 256) {
    float bias = b2[c];
    #pragma unroll
    for (int n = 0; n < NB; ++n) {
      float u = bias;
      #pragma unroll
      for (int m = 0; m < NB; ++m) u += scn[n * NB + m] * sz[m * FOUT + c];
      u = u * scale[n] + shift[n];
      u = (u >= 0.f) ? u : 0.2f * u;
      if (add) ob[n * FOUT + c] += u; else ob[n * FOUT + c] = u;
    }
  }
}

// ---------------------------------------------------------------------------
// Host-side orchestration.
// ---------------------------------------------------------------------------
extern "C" void kernel_launch(void* const* d_in, const int* in_sizes, int n_in,
                              void* d_out, int out_size, void* d_ws, size_t ws_size,
                              hipStream_t stream) {
  const float* x      = (const float*)d_in[0];
  const float* adj1   = (const float*)d_in[1];
  const float* adj2   = (const float*)d_in[2];
  const float* fc_w   = (const float*)d_in[3];
  const float* fc_b   = (const float*)d_in[4];
  const float* w1     = (const float*)d_in[5];
  const float* b1     = (const float*)d_in[6];
  const float* w2     = (const float*)d_in[7];
  const float* b2     = (const float*)d_in[8];
  const float* gamma1 = (const float*)d_in[9];
  const float* beta1  = (const float*)d_in[10];
  const float* gamma2 = (const float*)d_in[11];
  const float* beta2  = (const float*)d_in[12];
  const float* gamma3 = (const float*)d_in[13];
  const float* beta3  = (const float*)d_in[14];

  float* out = (float*)d_out;
  float* ws  = (float*)d_ws;

  // output layout: [sum(B*14*1024) | corr(B*196) | h(B*14*1024)]
  const size_t OUT_SUM  = 0;
  const size_t OUT_CORR = (size_t)BATCH * NB * FOUT;            // 58720256
  const size_t OUT_H    = OUT_CORR + (size_t)BATCH * NB * NB;   // 59523072

  // workspace layout (floats); total ~117.45M floats (~450 MB)
  const size_t WS_ADJ   = 0;                                    // 2*196
  const size_t WS_STATS = 512;                                  // gcn1: 56, gcn2: at +64
  const size_t WS_XW1   = 1024;
  const size_t XW1_SZ   = (size_t)BATCH * NB * HMID;            // 29360128
  const size_t WS_O     = WS_XW1 + XW1_SZ;
  const size_t WS_Z     = WS_O + XW1_SZ;

  float* adjn   = ws + WS_ADJ;
  float* st1    = ws + WS_STATS;        // s1,q1,s3,q3 (4*14)
  float* stu    = ws + WS_STATS + 64;   // su,qu (2*14)
  float* xw1    = ws + WS_XW1;
  float* obuf   = ws + WS_O;
  float* zbuf   = ws + WS_Z;

  dim3 blk(256);

  // 1) normalized adjacencies
  adjnorm_kernel<<<1, blk, 0, stream>>>(adj1, adj2, adjn);

  // 2) h = leaky(x @ fc_w + fc_b)   [WMMA GEMM, epi=1]
  gemm_bf16x3_kernel<<<dim3(FOUT / 128, MROWS / 128), blk, 0, stream>>>(
      x, fc_w, fc_b, out + OUT_H, MROWS, FIN, FOUT, 1);

  // 3) corr from h
  corr_kernel<<<BATCH, blk, 0, stream>>>(out + OUT_H, out + OUT_CORR);

  // 4) xw1 = x @ gcn1_w            [WMMA GEMM]
  gemm_bf16x3_kernel<<<dim3(HMID / 128, MROWS / 128), blk, 0, stream>>>(
      x, w1, nullptr, xw1, MROWS, FIN, HMID, 0);

  // 5) gcn1 BN stats (both adjacencies in one pass over xw1)
  hipMemsetAsync(st1, 0, 64 * sizeof(float), stream);
  gcn1_stats_kernel<<<BATCH, blk, 0, stream>>>(xw1, adjn, b1, st1);

  // 6) two gcn2 paths, sequential (shared o / z buffers)
  for (int path = 0; path < 2; ++path) {
    const float* adjn_p  = adjn + (path ? NB * NB : 0);
    const float* gamma_p = path ? gamma3 : gamma1;
    const float* beta_p  = path ? beta3  : beta1;
    const float* s_p     = st1 + (path ? 2 * NB : 0);
    const float* q_p     = st1 + (path ? 3 * NB : NB);

    // o = leaky(BN(adjn_p @ xw1 + b1))
    gcn1_apply_kernel<<<BATCH, blk, 0, stream>>>(
        xw1, adjn_p, b1, gamma_p, beta_p, s_p, q_p, obuf);

    // z = o @ gcn2_w               [WMMA GEMM]
    gemm_bf16x3_kernel<<<dim3(FOUT / 128, MROWS / 128), blk, 0, stream>>>(
        obuf, w2, nullptr, zbuf, MROWS, HMID, FOUT, 0);

    // gcn2 BN stats for u = corrn @ z + b2
    hipMemsetAsync(stu, 0, 32 * sizeof(float), stream);
    gcn2_stats_kernel<<<BATCH, blk, 0, stream>>>(zbuf, out + OUT_CORR, b2, stu);

    // out += leaky(BN(u))  (store on path 0, accumulate on path 1)
    gcn2_final_kernel<<<BATCH, blk, 0, stream>>>(
        zbuf, out + OUT_CORR, b2, gamma2, beta2, stu, out + OUT_SUM, path);
  }
}
